// StrippedEncoderBlock_78537771974785
// MI455X (gfx1250) — matrix operand order repaired and checked by reading, not compiled, not run on hardware
//
#include <hip/hip_runtime.h>

#define Bq 2
#define Hh 16
#define Sq 2048
#define Dd 64
#define BH (Bq * Hh)

typedef __bf16 bf16_t;
typedef bf16_t v16bf __attribute__((ext_vector_type(16)));
typedef float  v8f   __attribute__((ext_vector_type(8)));

#define KSTR 72    // padded K-tile row stride (bf16), 144B: 16B-aligned rows, conflict-free
#define VSTR 136   // padded transposed-V row stride, 272B
#define PSTR 40    // padded per-wave P scratch row stride, 80B

union Frag {
  uint4 u[2];
  v16bf v;
  bf16_t s[16];
};

__device__ __forceinline__ v8f wmma_bf16(v16bf a, v16bf b, v8f c) {
  return __builtin_amdgcn_wmma_f32_16x16x32_bf16(false, a, false, b, (short)0, c,
                                                 false, false);
}

// cross-lane reductions within 16-lane halves (rows are split lanes0-15/16-31)
__device__ __forceinline__ float redmax16(float x) {
  x = fmaxf(x, __shfl_xor(x, 1, 32));
  x = fmaxf(x, __shfl_xor(x, 2, 32));
  x = fmaxf(x, __shfl_xor(x, 4, 32));
  x = fmaxf(x, __shfl_xor(x, 8, 32));
  return x;
}
__device__ __forceinline__ float redsum16(float x) {
  x += __shfl_xor(x, 1, 32);
  x += __shfl_xor(x, 2, 32);
  x += __shfl_xor(x, 4, 32);
  x += __shfl_xor(x, 8, 32);
  return x;
}

// ---- prepass: K -> bf16 row-major, V -> bf16 transposed (d-major) ----------
__global__ __launch_bounds__(256)
void preconvert_kv(const float* __restrict__ K, const float* __restrict__ V,
                   bf16_t* __restrict__ Kb, bf16_t* __restrict__ Vtb) {
  __shared__ __attribute__((aligned(16))) bf16_t Lt[64 * KSTR];
  const int tid = threadIdx.x;
  const int bh  = blockIdx.y;
  const int k0  = blockIdx.x * 64;
  const size_t base = (size_t)bh * Sq * Dd;
#pragma unroll
  for (int j = 0; j < 4; ++j) {
    int c   = tid + j * 256;             // 64 keys x 16 float4
    int key = c >> 4;
    int d4  = (c & 15) << 2;
    size_t gi = base + (size_t)(k0 + key) * Dd + d4;
    float4 fk = *(const float4*)(K + gi);
    union { bf16_t h[4]; uint2 u; } pk;
    pk.h[0] = (bf16_t)fk.x; pk.h[1] = (bf16_t)fk.y;
    pk.h[2] = (bf16_t)fk.z; pk.h[3] = (bf16_t)fk.w;
    *(uint2*)(Kb + gi) = pk.u;
    float4 fv = *(const float4*)(V + gi);
    union { bf16_t h[4]; uint2 u; } pv;
    pv.h[0] = (bf16_t)fv.x; pv.h[1] = (bf16_t)fv.y;
    pv.h[2] = (bf16_t)fv.z; pv.h[3] = (bf16_t)fv.w;
    *(uint2*)&Lt[key * KSTR + d4] = pv.u;
  }
  __syncthreads();
  // transpose 64x64 tile out of LDS, coalesced 32B stores along keys
  const int d   = tid >> 2;
  const int k16 = (tid & 3) << 4;
  union { uint4 u[2]; bf16_t h[16]; } o;
#pragma unroll
  for (int i = 0; i < 16; ++i) o.h[i] = Lt[(k16 + i) * KSTR + d];
  bf16_t* dst = Vtb + (size_t)bh * Dd * Sq + (size_t)d * Sq + k0 + k16;
  *(uint4*)dst       = o.u[0];
  *((uint4*)dst + 1) = o.u[1];
}

// ---- main flash-attention kernel; PRE=1 uses bf16 workspace + async copies -
template <int PRE>
__global__ __launch_bounds__(256, 1)
void fa_fwd_bf16wmma(const float* __restrict__ Q, const float* __restrict__ K,
                     const float* __restrict__ V, const float* __restrict__ temp,
                     float* __restrict__ O, const bf16_t* __restrict__ Kb,
                     const bf16_t* __restrict__ Vtb) {
  __shared__ __attribute__((aligned(16))) bf16_t Kt[128 * KSTR]; // 18.0 KB
  __shared__ __attribute__((aligned(16))) bf16_t Vt[64 * VSTR];  // 17.0 KB (d-major)
  __shared__ __attribute__((aligned(16))) bf16_t Pl[8 * 16 * PSTR]; // 10 KB

  const int tid  = threadIdx.x;
  const int lane = tid & 31;
  const int wv   = tid >> 5;
  const int hlf  = lane >> 4;
  const int ln16 = lane & 15;

  const int bh = blockIdx.y;
  const int rb = blockIdx.x;
  const size_t base = (size_t)bh * Sq * Dd;

  const float cscale = 1.44269504088896340736f / (temp[0] * 8.0f);

  // Q tile 16x64 pre-scaled as two 16x32 bf16 A-fragments
  const int qrow = rb * 128 + wv * 16 + ln16;
  const float* qp = Q + base + (size_t)qrow * Dd;
  Frag qa[2];
#pragma unroll
  for (int f = 0; f < 2; ++f) {
    const float* gp = qp + f * 32 + hlf * 8;
    float4 a0 = *(const float4*)(gp);
    float4 a1 = *(const float4*)(gp + 4);
    float4 b0 = *(const float4*)(gp + 16);
    float4 b1 = *(const float4*)(gp + 20);
    qa[f].s[0] = (bf16_t)(a0.x * cscale);  qa[f].s[1] = (bf16_t)(a0.y * cscale);
    qa[f].s[2] = (bf16_t)(a0.z * cscale);  qa[f].s[3] = (bf16_t)(a0.w * cscale);
    qa[f].s[4] = (bf16_t)(a1.x * cscale);  qa[f].s[5] = (bf16_t)(a1.y * cscale);
    qa[f].s[6] = (bf16_t)(a1.z * cscale);  qa[f].s[7] = (bf16_t)(a1.w * cscale);
    qa[f].s[8]  = (bf16_t)(b0.x * cscale); qa[f].s[9]  = (bf16_t)(b0.y * cscale);
    qa[f].s[10] = (bf16_t)(b0.z * cscale); qa[f].s[11] = (bf16_t)(b0.w * cscale);
    qa[f].s[12] = (bf16_t)(b1.x * cscale); qa[f].s[13] = (bf16_t)(b1.y * cscale);
    qa[f].s[14] = (bf16_t)(b1.z * cscale); qa[f].s[15] = (bf16_t)(b1.w * cscale);
  }

  const v8f z = {0.f, 0.f, 0.f, 0.f, 0.f, 0.f, 0.f, 0.f};
  v8f acc[4];
#pragma unroll
  for (int t = 0; t < 4; ++t) acc[t] = z;
  float m_r[8], l_r[8];
#pragma unroll
  for (int r = 0; r < 8; ++r) { m_r[r] = -1e30f; l_r[r] = 0.f; }

  const float* kbG = K + base;
  const float* vbG = V + base;
  const unsigned long long kS = (unsigned long long)(Kb + base);
  const unsigned long long vS = (unsigned long long)(Vtb + (size_t)bh * Dd * Sq);

  for (int kb = 0; kb < Sq; kb += 128) {
    if (PRE) {
      // pure bf16 copy global->LDS via async DMA path (no VGPR round trip)
#pragma unroll
      for (int i = 0; i < 4; ++i) {
        int c   = tid + i * 256;          // 128 rows x 8 chunks of 8 bf16
        int key = c >> 3;
        int d8  = (c & 7) << 3;
        unsigned lds = (unsigned)(uintptr_t)&Kt[key * KSTR + d8];
        unsigned go  = (unsigned)(((kb + key) * Dd + d8) * 2);
        asm volatile("global_load_async_to_lds_b128 %0, %1, %2"
                     :: "v"(lds), "v"(go), "s"(kS) : "memory");
      }
#pragma unroll
      for (int i = 0; i < 4; ++i) {
        int c  = tid + i * 256;           // 64 d-rows x 16 chunks of 8 bf16
        int dd = c >> 4;
        int k8 = (c & 15) << 3;
        unsigned lds = (unsigned)(uintptr_t)&Vt[dd * VSTR + k8];
        unsigned go  = (unsigned)((dd * Sq + kb + k8) * 2);
        asm volatile("global_load_async_to_lds_b128 %0, %1, %2"
                     :: "v"(lds), "v"(go), "s"(vS) : "memory");
      }
      if (kb + 128 < Sq) {
        int prow = tid >> 1;
        __builtin_prefetch((const bf16_t*)kS + (kb + 128 + prow) * Dd + (tid & 1) * 32, 0, 1);
        __builtin_prefetch((const bf16_t*)vS + (tid & 127) * Sq + kb + 128 + (tid >> 7) * 64, 0, 1);
      }
      asm volatile("s_wait_asynccnt 0x0" ::: "memory");
    } else {
      // fallback: stage + convert fp32 -> bf16 in-kernel
#pragma unroll
      for (int it = 0; it < 8; ++it) {
        int c   = tid + it * 256;
        int key = c >> 4;
        int d4  = (c & 15) << 2;
        float4 fk = *(const float4*)(kbG + (size_t)(kb + key) * Dd + d4);
        union { bf16_t h[4]; uint2 u; } pk;
        pk.h[0] = (bf16_t)fk.x; pk.h[1] = (bf16_t)fk.y;
        pk.h[2] = (bf16_t)fk.z; pk.h[3] = (bf16_t)fk.w;
        *(uint2*)&Kt[key * KSTR + d4] = pk.u;
        float4 fv = *(const float4*)(vbG + (size_t)(kb + key) * Dd + d4);
        Vt[(d4 + 0) * VSTR + key] = (bf16_t)fv.x;
        Vt[(d4 + 1) * VSTR + key] = (bf16_t)fv.y;
        Vt[(d4 + 2) * VSTR + key] = (bf16_t)fv.z;
        Vt[(d4 + 3) * VSTR + key] = (bf16_t)fv.w;
      }
      if (kb + 128 < Sq) {
        int prow = tid >> 1;
        int pcol = (tid & 1) << 5;
        __builtin_prefetch(kbG + (size_t)(kb + 128 + prow) * Dd + pcol, 0, 1);
        __builtin_prefetch(vbG + (size_t)(kb + 128 + prow) * Dd + pcol, 0, 1);
      }
    }
    __syncthreads();

    // ---- all 8 score tiles (16 WMMA) ---------------------------------------
    v8f sc[8];
#pragma unroll
    for (int n = 0; n < 8; ++n) {
      const bf16_t* kr = &Kt[(n * 16 + ln16) * KSTR + hlf * 8];
      Frag f0, f1;
      f0.u[0] = *(const uint4*)(kr);      f0.u[1] = *(const uint4*)(kr + 16);
      f1.u[0] = *(const uint4*)(kr + 32); f1.u[1] = *(const uint4*)(kr + 48);
      v8f s = wmma_bf16(qa[0].v, f0.v, z);
      sc[n] = wmma_bf16(qa[1].v, f1.v, s);
    }

    // ---- deferred online softmax, one update per 128 keys ------------------
    float alpha[8];
#pragma unroll
    for (int r = 0; r < 8; ++r) {
      float mx = sc[0][r];
#pragma unroll
      for (int n = 1; n < 8; ++n) mx = fmaxf(mx, sc[n][r]);
      mx = redmax16(mx);
      float mn = fmaxf(m_r[r], mx);
      alpha[r] = __builtin_amdgcn_exp2f(m_r[r] - mn);
      m_r[r]   = mn;
      float ls = 0.f;
#pragma unroll
      for (int n = 0; n < 8; ++n) {
        float p = __builtin_amdgcn_exp2f(sc[n][r] - mn);
        sc[n][r] = p;
        ls += p;
      }
      l_r[r] = l_r[r] * alpha[r] + ls;   // per-lane partial; reduced once at end
    }
#pragma unroll
    for (int t = 0; t < 4; ++t)
#pragma unroll
      for (int r = 0; r < 8; ++r) acc[t][r] *= alpha[r];

    // ---- P x V over 4 chunks of 32 keys ------------------------------------
#pragma unroll
    for (int kc = 0; kc < 4; ++kc) {
      const int ck = kc * 32;
      bf16_t* pw = &Pl[wv * 16 * PSTR];
#pragma unroll
      for (int r = 0; r < 8; ++r) {
        int row = r + hlf * 8;
        pw[row * PSTR + ln16]      = (bf16_t)sc[2 * kc][r];
        pw[row * PSTR + 16 + ln16] = (bf16_t)sc[2 * kc + 1][r];
      }
      Frag pf;
      pf.u[0] = *(const uint4*)&pw[ln16 * PSTR + hlf * 8];
      pf.u[1] = *(const uint4*)&pw[ln16 * PSTR + 16 + hlf * 8];
#pragma unroll
      for (int t = 0; t < 4; ++t) {
        const bf16_t* vr = &Vt[(t * 16 + ln16) * VSTR + ck + hlf * 8];
        Frag vf;
        vf.u[0] = *(const uint4*)(vr);
        vf.u[1] = *(const uint4*)(vr + 16);
        acc[t] = wmma_bf16(pf.v, vf.v, acc[t]);
      }
    }
    __syncthreads();
  }

  // ---- epilogue -------------------------------------------------------------
  float* op = O + base + (size_t)(rb * 128 + wv * 16) * Dd;
#pragma unroll
  for (int r = 0; r < 8; ++r) {
    int row   = r + hlf * 8;
    float inv = 1.0f / redsum16(l_r[r]);
#pragma unroll
    for (int t = 0; t < 4; ++t)
      op[row * Dd + t * 16 + ln16] = acc[t][r] * inv;
  }
}

extern "C" void kernel_launch(void* const* d_in, const int* in_sizes, int n_in,
                              void* d_out, int out_size, void* d_ws, size_t ws_size,
                              hipStream_t stream) {
  (void)in_sizes; (void)n_in; (void)out_size;
  const float* q    = (const float*)d_in[0];
  const float* k    = (const float*)d_in[1];
  const float* v    = (const float*)d_in[2];
  const float* temp = (const float*)d_in[3];
  float* out = (float*)d_out;

  const size_t kv_elems = (size_t)BH * Sq * Dd;          // 4,194,304 per tensor
  const size_t need     = 2 * kv_elems * sizeof(bf16_t); // 16.8 MB
  dim3 grid(Sq / 128, BH);

  if (d_ws && ws_size >= need) {
    bf16_t* Kb  = (bf16_t*)d_ws;
    bf16_t* Vtb = Kb + kv_elems;
    preconvert_kv<<<dim3(Sq / 64, BH), dim3(256), 0, stream>>>(k, v, Kb, Vtb);
    fa_fwd_bf16wmma<1><<<grid, dim3(256), 0, stream>>>(q, k, v, temp, out, Kb, Vtb);
  } else {
    fa_fwd_bf16wmma<0><<<grid, dim3(256), 0, stream>>>(q, k, v, temp, out,
                                                       nullptr, nullptr);
  }
}
